// FlightDelayGNN_26542897890030
// MI455X (gfx1250) — compile-verified, matrix operand order checked by reading, not run on hardware
//
#include <hip/hip_runtime.h>
#include <math.h>

#define N_APT 400
#define N_FLT 60000
#define AP_IN 32
#define FL_IN 48
#define HIDN 256
#define E_AA 15000
#define E_ROT 50000
#define NTOT (N_APT + N_FLT)
#define E_TOT (3 * E_AA + 4 * N_FLT)

typedef __attribute__((ext_vector_type(16))) __bf16 v16bf;
typedef __attribute__((ext_vector_type(8)))  float  v8f;

union BFrag { v16bf v; uint4 q[2]; unsigned short u[16]; };

__device__ inline unsigned short f2bf(float f) {
  unsigned u = __float_as_uint(f);
  u += 0x7FFFu + ((u >> 16) & 1u);           // round-to-nearest-even truncation
  return (unsigned short)(u >> 16);
}
__device__ inline unsigned pk2(float a, float b) {
  return (unsigned)f2bf(a) | ((unsigned)f2bf(b) << 16);
}
__device__ inline float sigm(float x) { return 1.f / (1.f + expf(-x)); }
__device__ inline void atomAddF(float* p, float v) {
  __hip_atomic_fetch_add(p, v, __ATOMIC_RELAXED, __HIP_MEMORY_SCOPE_AGENT);
}
__device__ inline void atomMaxU(unsigned* p, unsigned v) {
  __hip_atomic_fetch_max(p, v, __ATOMIC_RELAXED, __HIP_MEMORY_SCOPE_AGENT);
}
__device__ inline unsigned ffl(float f) {
  unsigned u = __float_as_uint(f);
  return (u & 0x80000000u) ? ~u : (u | 0x80000000u);
}
__device__ inline float funfl(unsigned u) {
  return __uint_as_float((u & 0x80000000u) ? (u & 0x7FFFFFFFu) : ~u);
}
__device__ inline float waveSum(float s) {
  #pragma unroll
  for (int off = 16; off > 0; off >>= 1) s += __shfl_xor(s, off, 32);
  return s;
}

// ---------------------------------------------------------------------------
// WMMA bf16 GEMM: C[M,N] = A[M,K] x B (+bias); optional transB, fused relu,
// accumulate-into-C, batched strides (grid.z).
// Block tile 128x64x32, 128 threads = 4 waves; each wave owns a 32x64 sub-tile
// (2 A-fragments x 4 B-fragments = 8 WMMAs / k-step, B fragments reused).
// Staging: registers -> packed bf16 pairs -> uint4 ds_store_b128 only.
// Non-transposed B addressing uses 32-bit offsets (saddr + voffset form, less VALU).
// ---------------------------------------------------------------------------
__global__ __launch_bounds__(128)
void gemm_wmma_k(const float* __restrict__ A, const float* __restrict__ B,
                 const float* __restrict__ bias, float* __restrict__ C,
                 int M, int N, int K, int lda, int ldb, int ldc,
                 long long aB, long long bB, long long cB, long long biasB,
                 int transB, int reluAct, int accum)
{
  __shared__ __align__(16) unsigned short As[128][40];  // [m][k], +8 pad
  __shared__ __align__(16) unsigned short Bs[64][40];   // [n][k], +8 pad
  const int bz = blockIdx.z;
  A += (long long)bz * aB;
  B += (long long)bz * bB;
  C += (long long)bz * cB;
  if (bias) bias += (long long)bz * biasB;

  const int m0 = blockIdx.y * 128;
  const int n0 = blockIdx.x * 64;
  const int tid = threadIdx.x;
  const int wave = tid >> 5;
  const int lane = tid & 31;
  const int lrow = lane & 15;
  const int lhalf = lane >> 4;

  v8f acc[2][4];
  #pragma unroll
  for (int a = 0; a < 2; a++)
    #pragma unroll
    for (int i = 0; i < 4; i++)
      #pragma unroll
      for (int j = 0; j < 8; j++) acc[a][i][j] = 0.f;

  for (int k0 = 0; k0 < K; k0 += 32) {
    { // ---- stage A: thread t owns row t, all 32 k ----
      const int gm = m0 + tid;
      float fa[32];
      if (gm < M && (k0 + 32) <= K) {
        const float4* src = (const float4*)(A + (long long)gm * lda + k0);
        #pragma unroll
        for (int i = 0; i < 8; i++) {
          float4 f = src[i];
          fa[4 * i + 0] = f.x; fa[4 * i + 1] = f.y;
          fa[4 * i + 2] = f.z; fa[4 * i + 3] = f.w;
        }
      } else {
        #pragma unroll
        for (int i = 0; i < 32; i++) {
          int gk = k0 + i;
          fa[i] = (gm < M && gk < K) ? A[(long long)gm * lda + gk] : 0.f;
        }
      }
      uint4* dst = (uint4*)&As[tid][0];
      #pragma unroll
      for (int i = 0; i < 4; i++) {
        uint4 p;
        p.x = pk2(fa[8 * i + 0], fa[8 * i + 1]);
        p.y = pk2(fa[8 * i + 2], fa[8 * i + 3]);
        p.z = pk2(fa[8 * i + 4], fa[8 * i + 5]);
        p.w = pk2(fa[8 * i + 6], fa[8 * i + 7]);
        dst[i] = p;
      }
    }
    { // ---- stage B transposed into Bs[n][k]: thread owns 16 k of one n ----
      const int bn = tid >> 1;
      const int ks = (tid & 1) * 16;
      const int gn = n0 + bn;
      float fv[16];
      if (transB) {
        if (gn < N && (k0 + ks + 16) <= K) {
          const float4* src = (const float4*)(B + (long long)gn * ldb + k0 + ks);
          #pragma unroll
          for (int i = 0; i < 4; i++) {
            float4 f = src[i];
            fv[4 * i + 0] = f.x; fv[4 * i + 1] = f.y;
            fv[4 * i + 2] = f.z; fv[4 * i + 3] = f.w;
          }
        } else {
          #pragma unroll
          for (int i = 0; i < 16; i++) {
            int gk = k0 + ks + i;
            fv[i] = (gn < N && gk < K) ? B[(long long)gn * ldb + gk] : 0.f;
          }
        }
      } else {
        // 32-bit offsets: uniform (k0*ldb, i*ldb) parts stay scalar, gn and ks*ldb are the
        // per-thread terms -> saddr + voffset addressing, no u64 adds.
        const unsigned base = (unsigned)((k0 + ks) * ldb + gn);
        if (gn < N && (k0 + ks + 16) <= K) {
          #pragma unroll
          for (int i = 0; i < 16; i++)
            fv[i] = B[base + (unsigned)(i * ldb)];
        } else {
          #pragma unroll
          for (int i = 0; i < 16; i++) {
            int gk = k0 + ks + i;
            fv[i] = (gn < N && gk < K) ? B[base + (unsigned)(i * ldb)] : 0.f;
          }
        }
      }
      uint4* dst = (uint4*)&Bs[bn][ks];
      #pragma unroll
      for (int i = 0; i < 2; i++) {
        uint4 p;
        p.x = pk2(fv[8 * i + 0], fv[8 * i + 1]);
        p.y = pk2(fv[8 * i + 2], fv[8 * i + 3]);
        p.z = pk2(fv[8 * i + 4], fv[8 * i + 5]);
        p.w = pk2(fv[8 * i + 6], fv[8 * i + 7]);
        dst[i] = p;
      }
    }
    __syncthreads();

    // Fragments: 16-bit A 16x32 ISA layout — elems 0..7 = K lhalf*8+0..7,
    // elems 8..15 = K 16+lhalf*8+0..7; lane row = lrow.
    BFrag af[2];
    #pragma unroll
    for (int r2 = 0; r2 < 2; r2++) {
      const unsigned short* arow = &As[wave * 32 + r2 * 16 + lrow][0];
      af[r2].q[0] = *(const uint4*)(arow + lhalf * 8);
      af[r2].q[1] = *(const uint4*)(arow + 16 + lhalf * 8);
    }
    #pragma unroll
    for (int nf = 0; nf < 4; nf++) {
      BFrag bfr;
      const unsigned short* brow = &Bs[nf * 16 + lrow][0];
      bfr.q[0] = *(const uint4*)(brow + lhalf * 8);
      bfr.q[1] = *(const uint4*)(brow + 16 + lhalf * 8);
      acc[0][nf] = __builtin_amdgcn_wmma_f32_16x16x32_bf16(
          false, af[0].v, false, bfr.v, (short)0, acc[0][nf], false, false);
      acc[1][nf] = __builtin_amdgcn_wmma_f32_16x16x32_bf16(
          false, af[1].v, false, bfr.v, (short)0, acc[1][nf], false, false);
    }
    __syncthreads();
  }

  // C layout: elem r -> row = base + lhalf*8 + r, col = n0 + nf*16 + lrow
  #pragma unroll
  for (int r2 = 0; r2 < 2; r2++) {
    #pragma unroll
    for (int nf = 0; nf < 4; nf++) {
      const int col = n0 + nf * 16 + lrow;
      if (col >= N) continue;
      const float bv = bias ? bias[col] : 0.f;
      #pragma unroll
      for (int r = 0; r < 8; r++) {
        const int row = m0 + wave * 32 + r2 * 16 + lhalf * 8 + r;
        if (row >= M) continue;
        float v = acc[r2][nf][r] + bv;
        long long idx = (long long)row * ldc + col;
        if (accum) v += C[idx];
        if (reluAct) v = fmaxf(v, 0.f);
        C[idx] = v;
      }
    }
  }
}

// ---------------------------------------------------------------------------
// Elementwise / reduction kernels
// ---------------------------------------------------------------------------
__global__ void fill_f32_k(float* p, float v, long long n) {
  long long i = blockIdx.x * (long long)blockDim.x + threadIdx.x;
  long long st = (long long)gridDim.x * blockDim.x;
  for (; i < n; i += st) p[i] = v;
}

__global__ __launch_bounds__(256)
void ln256_k(float* __restrict__ x, const float* __restrict__ g,
             const float* __restrict__ b, int N, int reluAct)
{
  int row = blockIdx.x * 8 + (threadIdx.x >> 5);
  int lane = threadIdx.x & 31;
  if (row >= N) return;
  float* xr = x + (long long)row * 256;
  float v[8]; float s = 0.f;
  #pragma unroll
  for (int i = 0; i < 8; i++) { v[i] = xr[lane + i * 32]; s += v[i]; }
  s = waveSum(s);
  float mean = s * (1.f / 256.f);
  float vs = 0.f;
  #pragma unroll
  for (int i = 0; i < 8; i++) { float d = v[i] - mean; vs += d * d; }
  vs = waveSum(vs);
  float inv = rsqrtf(vs * (1.f / 256.f) + 1e-5f);
  #pragma unroll
  for (int i = 0; i < 8; i++) {
    int c = lane + i * 32;
    float y = (v[i] - mean) * inv * g[c] + b[c];
    if (reluAct) y = fmaxf(y, 0.f);
    xr[c] = y;
  }
}

// x := LN( sk*o + (2-sk)*x )  with sk = sigmoid(*skipP)   (res + residual)
__global__ __launch_bounds__(256)
void ln256_blend_k(const float* __restrict__ o, float* __restrict__ x,
                   const float* __restrict__ skipP, const float* __restrict__ g,
                   const float* __restrict__ b, int N)
{
  int row = blockIdx.x * 8 + (threadIdx.x >> 5);
  int lane = threadIdx.x & 31;
  if (row >= N) return;
  float sk = sigm(skipP[0]);
  const float* orow = o + (long long)row * 256;
  float* xr = x + (long long)row * 256;
  float v[8]; float s = 0.f;
  #pragma unroll
  for (int i = 0; i < 8; i++) {
    int c = lane + i * 32;
    v[i] = sk * orow[c] + (2.f - sk) * xr[c];
    s += v[i];
  }
  s = waveSum(s);
  float mean = s * (1.f / 256.f);
  float vs = 0.f;
  #pragma unroll
  for (int i = 0; i < 8; i++) { float d = v[i] - mean; vs += d * d; }
  vs = waveSum(vs);
  float inv = rsqrtf(vs * (1.f / 256.f) + 1e-5f);
  #pragma unroll
  for (int i = 0; i < 8; i++) {
    int c = lane + i * 32;
    xr[c] = (v[i] - mean) * inv * g[c] + b[c];
  }
}

__global__ void gelu_k(float* p, long long n) {
  long long i = blockIdx.x * (long long)blockDim.x + threadIdx.x;
  if (i >= n) return;
  float x = p[i];
  p[i] = 0.5f * x * (1.f + erff(x * 0.70710678118654752f));
}
__global__ void sigmoid_k(float* p, long long n) {
  long long i = blockIdx.x * (long long)blockDim.x + threadIdx.x;
  if (i >= n) return;
  p[i] = sigm(p[i]);
}
__global__ void fuse_out_k(const float* gate, const float* fuse, const float* emb,
                           float* out, long long n) {
  long long i = blockIdx.x * (long long)blockDim.x + threadIdx.x;
  if (i >= n) return;
  out[i] = gate[i] * fuse[i] + emb[i];
}

__global__ void rotgate_k(const float* __restrict__ attr, const float* __restrict__ W1,
                          const float* __restrict__ b1, const float* __restrict__ W2,
                          const float* __restrict__ b2, float* __restrict__ g, int E)
{
  int e = blockIdx.x * blockDim.x + threadIdx.x;
  if (e >= E) return;
  float a0 = attr[2 * e], a1 = attr[2 * e + 1];
  float acc = b2[0];
  #pragma unroll
  for (int j = 0; j < 32; j++) {
    float h = fmaxf(a0 * W1[j] + a1 * W1[32 + j] + b1[j], 0.f);
    acc += h * W2[j];
  }
  g[e] = sigm(acc);
}
__global__ void rot_scatter_k(float* __restrict__ xfl, const float* __restrict__ x0,
                              const float* __restrict__ g, const int* __restrict__ src,
                              const int* __restrict__ dst, long long n)
{
  long long i = blockIdx.x * (long long)blockDim.x + threadIdx.x;
  if (i >= n) return;
  int e = (int)(i >> 8), c = (int)(i & 255);
  atomAddF(&xfl[(long long)dst[e] * 256 + c], g[e] * x0[(long long)src[e] * 256 + c]);
}

__global__ void edge_logit_k(const float* __restrict__ kqvD, const float* __restrict__ kt,
                             const int* __restrict__ src, const int* __restrict__ dst,
                             int E, int dstOff, const float* __restrict__ prel,
                             unsigned* __restrict__ amax, float* __restrict__ aout)
{
  int idx = blockIdx.x * blockDim.x + threadIdx.x;
  if (idx >= E * 8) return;
  int e = idx >> 3, h = idx & 7;
  int sN = src[e], dN = dst[e];
  const float* q = kqvD + (long long)dN * 768 + 256 + h * 32;
  const float* k = kt + (long long)sN * 256 + h * 32;
  float a = 0.f;
  #pragma unroll
  for (int i = 0; i < 32; i++) a += q[i] * k[i];
  a *= prel[h] * 0.17677669529663689f;   // 1/sqrt(D)
  aout[idx] = a;
  atomMaxU(&amax[(long long)(dN + dstOff) * 8 + h], ffl(a));
}

__global__ void edge_soft_k(float* __restrict__ aout, const int* __restrict__ dst,
                            int E, int dstOff, const unsigned* __restrict__ amax,
                            float* __restrict__ denom)
{
  int idx = blockIdx.x * blockDim.x + threadIdx.x;
  if (idx >= E * 8) return;
  int e = idx >> 3, h = idx & 7;
  long long seg = (long long)(dst[e] + dstOff) * 8 + h;
  float ae = expf(aout[idx] - funfl(amax[seg]));
  aout[idx] = ae;
  atomAddF(&denom[seg], ae);
}

__global__ void edge_accum_k(const float* __restrict__ aout, const float* __restrict__ denom,
                             const float* __restrict__ vt, const int* __restrict__ src,
                             const int* __restrict__ dst, int E, int dstOff,
                             float* __restrict__ out)
{
  long long i = blockIdx.x * (long long)blockDim.x + threadIdx.x;
  if (i >= (long long)E * 256) return;
  int e = (int)(i >> 8), c = (int)(i & 255), h = c >> 5;
  int sN = src[e], dN = dst[e];
  float w = aout[(long long)e * 8 + h] / denom[(long long)(dN + dstOff) * 8 + h];
  atomAddF(&out[(long long)(dN + dstOff) * 256 + c], w * vt[(long long)sN * 256 + c]);
}

__global__ void gru_gate_k(const float* __restrict__ gi, const float* __restrict__ gh,
                           const float* __restrict__ h, float* __restrict__ out,
                           long long total, int H)
{
  long long i = blockIdx.x * (long long)blockDim.x + threadIdx.x;
  if (i >= total) return;
  long long n = i / H; int j = (int)(i % H);
  const float* a = gi + n * 3ll * H;
  const float* c = gh + n * 3ll * H;
  float r  = sigm(a[j] + c[j]);
  float z  = sigm(a[H + j] + c[H + j]);
  float nn = tanhf(a[2 * H + j] + r * c[2 * H + j]);
  out[i] = (1.f - z) * nn + z * h[i];
}

__global__ void gather_rows_k(const float* __restrict__ table, const int* __restrict__ ids,
                              float* __restrict__ out, long long n, int H)
{
  long long i = blockIdx.x * (long long)blockDim.x + threadIdx.x;
  if (i >= n) return;
  long long row = i / H; int c = (int)(i % H);
  out[i] = table[(long long)ids[row] * H + c];
}

__global__ void scatter_set_k(const int* __restrict__ tgt, const int* __restrict__ srcIdx,
                              const float* __restrict__ in, float* __restrict__ out,
                              long long n)
{
  long long i = blockIdx.x * (long long)blockDim.x + threadIdx.x;
  if (i >= n) return;
  int e = (int)(i >> 8), c = (int)(i & 255);
  out[(long long)tgt[e] * 256 + c] = in[(long long)srcIdx[e] * 256 + c];
}

__global__ void final_k(const float* __restrict__ zb, const float* __restrict__ logits,
                        const float* __restrict__ mag, float* __restrict__ pred,
                        float* __restrict__ zout, int N)
{
  int n = blockIdx.x * blockDim.x + threadIdx.x;
  if (n >= N) return;
  float z = zb[n];
  float p4 = sigm(logits[n * 6 + 4]);
  float p5 = sigm(logits[n * 6 + 5]);
  float m0r = mag[n * 2], m1r = mag[n * 2 + 1];
  float m0 = (m0r > 20.f) ? m0r : log1pf(expf(m0r));
  float m1 = (m1r > 20.f) ? m1r : log1pf(expf(m1r));
  z += p4 * m0 + p5 * m1;
  zout[n] = z;
  float sg = (z > 0.f) ? 1.f : ((z < 0.f) ? -1.f : 0.f);
  pred[n] = sg * expm1f(fabsf(z));
}

// ---------------------------------------------------------------------------
// Host orchestration
// ---------------------------------------------------------------------------
struct Bump {
  char* base; size_t off; size_t cap;
  float* f(size_t elems) {
    size_t a = (off + 255) & ~(size_t)255;
    off = a + elems * sizeof(float);
    return (float*)(base + a);
  }
  unsigned* u(size_t elems) { return (unsigned*)f(elems); }
};

static inline unsigned nblk(long long n) { return (unsigned)((n + 255) / 256); }

static void gemm(hipStream_t s, const float* A, const float* B, const float* bias, float* C,
                 int M, int N, int K, int lda, int ldb, int ldc,
                 int transB = 0, int relu = 0, int accum = 0, int batch = 1,
                 long long aB = 0, long long bB = 0, long long cB = 0, long long biasB = 0)
{
  dim3 g((N + 63) / 64, (M + 127) / 128, batch);
  gemm_wmma_k<<<g, dim3(128), 0, s>>>(A, B, bias, C, M, N, K, lda, ldb, ldc,
                                      aB, bB, cB, biasB, transB, relu, accum);
}

extern "C" void kernel_launch(void* const* d_in, const int* in_sizes, int n_in,
                              void* d_out, int out_size, void* d_ws, size_t ws_size,
                              hipStream_t stream)
{
  (void)in_sizes; (void)n_in; (void)out_size;
  hipStream_t s = stream;
  #define PRM(i) ((const float*)d_in[(i)])
  const float* ap_x = PRM(0);
  const float* fl_x = PRM(1);
  const float* rot_attr = PRM(2);
  const int* rot_src = (const int*)d_in[3];
  const int* rot_dst = (const int*)d_in[4];
  const int* esrc[7]; const int* edst[7];
  for (int t = 0; t < 7; t++) { esrc[t] = (const int*)d_in[5 + 2 * t]; edst[t] = (const int*)d_in[6 + 2 * t]; }
  const int* tail_id = (const int*)d_in[19];
  const float* tail_states = PRM(20);
  const float* ap_h0 = PRM(21);   // [1,400,256]

  const int Ecnt[7]  = {E_AA, E_AA, E_AA, N_FLT, N_FLT, N_FLT, N_FLT};
  const int eiOf[7]  = {0, 1, 2, 3, 4, 6, 7};
  const int srcFl[7] = {0, 0, 0, 0, 0, 1, 1};
  const int dstFl[7] = {0, 0, 0, 1, 1, 0, 0};
  long long eBase[7]; { long long a = 0; for (int t = 0; t < 7; t++) { eBase[t] = a; a += (long long)Ecnt[t] * 8; } }

  float* outp = (float*)d_out;
  float* o_appred = outp;                 // 400
  float* o_flpred = outp + 400;           // 60000
  float* o_flz    = outp + 60400;         // 60000
  float* o_logits = outp + 120400;        // 60000*6
  float* o_h1     = outp + 480400;        // 400*256 (ap_h_new)
  float* o_tail   = outp + 582800;        // 60000*128 (new_tail)

  Bump ws{(char*)d_ws, 0, ws_size};
  float* x_ap = ws.f((size_t)N_APT * 256);
  float* x_fl = ws.f((size_t)N_FLT * 256);
  size_t mark = ws.off;

  // ---- Phase 1: projections + LN + rotation-gated scatter ----
  gemm(s, ap_x, PRM(22), PRM(23), x_ap, N_APT, 256, 32, 32, 256, 256);
  ln256_k<<<(N_APT + 7) / 8, 256, 0, s>>>(x_ap, PRM(24), PRM(25), N_APT, 0);
  gemm(s, fl_x, PRM(26), PRM(27), x_fl, N_FLT, 256, 48, 48, 256, 256);
  ln256_k<<<(N_FLT + 7) / 8, 256, 0, s>>>(x_fl, PRM(28), PRM(29), N_FLT, 0);

  {
    ws.off = mark;
    float* xfl0 = ws.f((size_t)N_FLT * 256);
    float* grot = ws.f(E_ROT);
    hipMemcpyAsync(xfl0, x_fl, (size_t)N_FLT * 256 * 4, hipMemcpyDeviceToDevice, s);
    rotgate_k<<<nblk(E_ROT), 256, 0, s>>>(rot_attr, PRM(30), PRM(31), PRM(32), PRM(33), grot, E_ROT);
    long long n = (long long)E_ROT * 256;
    rot_scatter_k<<<nblk(n), 256, 0, s>>>(x_fl, xfl0, grot, rot_src, rot_dst, n);
  }

  // ---- Phase 2: 2x HGT conv layers ----
  ws.off = mark;
  float* kqv_ap = ws.f((size_t)N_APT * 768);
  float* kqv_fl = ws.f((size_t)N_FLT * 768);
  float* kt_ap  = ws.f((size_t)N_APT * 256);
  float* kt_fl  = ws.f((size_t)N_FLT * 256);
  float* vt_ap  = ws.f((size_t)5 * N_APT * 256);
  float* vt_fl  = ws.f((size_t)2 * N_FLT * 256);
  float* a_all  = ws.f((size_t)E_TOT * 8);
  unsigned* amax = ws.u((size_t)NTOT * 8);
  float* denom  = ws.f((size_t)NTOT * 8);
  float* attn   = ws.f((size_t)NTOT * 256);
  float* oc_ap  = ws.f((size_t)N_APT * 256);
  float* oc_fl  = ws.f((size_t)N_FLT * 256);

  for (int l = 0; l < 2; l++) {
    const int LB = 34 + l * 19;
    #define LP(i) PRM(LB + (i))
    gemm(s, x_ap, LP(0), LP(1), kqv_ap, N_APT, 768, 256, 256, 768, 768);
    gemm(s, x_fl, LP(2), LP(3), kqv_fl, N_FLT, 768, 256, 256, 768, 768);
    fill_f32_k<<<512, 256, 0, s>>>((float*)amax, 0.f, (long long)NTOT * 8);
    fill_f32_k<<<512, 256, 0, s>>>(denom, 0.f, (long long)NTOT * 8);
    fill_f32_k<<<4096, 256, 0, s>>>(attn, 0.f, (long long)NTOT * 256);

    float* vtp[7];
    int apS = 0, flS = 0;
    for (int t = 0; t < 7; t++) {
      int ei = eiOf[t];
      const float* kqvS = srcFl[t] ? kqv_fl : kqv_ap;
      int Ns = srcFl[t] ? N_FLT : N_APT;
      float* kt = srcFl[t] ? kt_fl : kt_ap;
      float* vt = srcFl[t] ? vt_fl + (size_t)flS * N_FLT * 256
                           : vt_ap + (size_t)apS * N_APT * 256;
      vtp[t] = vt;
      const float* Wk = LP(4) + (size_t)ei * 8192;
      const float* bk = LP(5) + (size_t)ei * 256;
      const float* Wv = LP(6) + (size_t)ei * 8192;
      const float* bv = LP(7) + (size_t)ei * 256;
      // einsum nhd,hde->nhe as batch-8 GEMM (A col offset h*32 inside lda=768)
      gemm(s, kqvS,       Wk, bk, kt, Ns, 32, 32, 768, 32, 256, 0, 0, 0, 8, 32, 1024, 32, 32);
      gemm(s, kqvS + 512, Wv, bv, vt, Ns, 32, 32, 768, 32, 256, 0, 0, 0, 8, 32, 1024, 32, 32);
      const float* kqvD = dstFl[t] ? kqv_fl : kqv_ap;
      int doff = dstFl[t] ? N_APT : 0;
      int E = Ecnt[t];
      edge_logit_k<<<nblk((long long)E * 8), 256, 0, s>>>(
          kqvD, kt, esrc[t], edst[t], E, doff, LP(8) + ei * 8, amax, a_all + eBase[t]);
      if (srcFl[t]) flS++; else apS++;
    }
    for (int t = 0; t < 7; t++) {
      int E = Ecnt[t], doff = dstFl[t] ? N_APT : 0;
      edge_soft_k<<<nblk((long long)E * 8), 256, 0, s>>>(a_all + eBase[t], edst[t], E, doff, amax, denom);
    }
    for (int t = 0; t < 7; t++) {
      int E = Ecnt[t], doff = dstFl[t] ? N_APT : 0;
      long long n = (long long)E * 256;
      edge_accum_k<<<nblk(n), 256, 0, s>>>(a_all + eBase[t], denom, vtp[t], esrc[t], edst[t], E, doff, attn);
    }
    gelu_k<<<nblk((long long)NTOT * 256), 256, 0, s>>>(attn, (long long)NTOT * 256);
    gemm(s, attn,                       LP(9),  LP(10), oc_ap, N_APT, 256, 256, 256, 256, 256);
    gemm(s, attn + (size_t)N_APT * 256, LP(11), LP(12), oc_fl, N_FLT, 256, 256, 256, 256, 256);
    ln256_blend_k<<<(N_APT + 7) / 8, 256, 0, s>>>(oc_ap, x_ap, LP(13), LP(15), LP(16), N_APT);
    ln256_blend_k<<<(N_FLT + 7) / 8, 256, 0, s>>>(oc_fl, x_fl, LP(14), LP(17), LP(18), N_FLT);
    #undef LP
  }

  // ---- Phase 3: GRUs, fusion, heads ----
  ws.off = mark;
  float* gi_ap   = ws.f((size_t)N_APT * 768);
  float* gh_ap   = ws.f((size_t)N_APT * 768);
  float* t1_ap   = ws.f((size_t)N_APT * 256);
  float* t2_ap   = ws.f((size_t)N_APT * 128);
  float* ctxall  = ws.f((size_t)N_APT * 256);
  float* apctx   = ws.f((size_t)N_FLT * 256);
  float* prevh   = ws.f((size_t)N_FLT * 128);
  float* gi_t    = ws.f((size_t)N_FLT * 384);
  float* gh_t    = ws.f((size_t)N_FLT * 384);
  float* tailctx = ws.f((size_t)N_FLT * 256);
  float* ffacc   = ws.f((size_t)N_FLT * 256);
  float* fgacc   = ws.f((size_t)N_FLT * 256);
  float* flout   = ws.f((size_t)N_FLT * 256);
  float* f1      = ws.f((size_t)N_FLT * 256);
  float* f2      = ws.f((size_t)N_FLT * 128);
  float* zb      = ws.f((size_t)N_FLT);
  float* mag     = ws.f((size_t)N_FLT * 2);

  // airport GRU -> h1 (written straight into ap_h_new output slot)
  gemm(s, x_ap,  PRM(72), PRM(74), gi_ap, N_APT, 768, 256, 256, 256, 768, 1);
  gemm(s, ap_h0, PRM(73), PRM(75), gh_ap, N_APT, 768, 256, 256, 256, 768, 1);
  gru_gate_k<<<nblk((long long)N_APT * 256), 256, 0, s>>>(gi_ap, gh_ap, ap_h0, o_h1, (long long)N_APT * 256, 256);
  // ap head -> ap_pred
  gemm(s, o_h1,  PRM(90), PRM(91), t1_ap, N_APT, 256, 256, 256, 256, 256, 0, 1);
  gemm(s, t1_ap, PRM(92), PRM(93), t2_ap, N_APT, 128, 256, 256, 128, 128, 0, 1);
  gemm(s, t2_ap, PRM(94), PRM(95), o_appred, N_APT, 1, 128, 128, 1, 1);
  // airport context scatter to flights
  gemm(s, o_h1, PRM(76), PRM(77), ctxall, N_APT, 256, 256, 256, 256, 256);
  fill_f32_k<<<4096, 256, 0, s>>>(apctx, 0.f, (long long)N_FLT * 256);
  {
    long long n = (long long)N_FLT * 256;
    scatter_set_k<<<nblk(n), 256, 0, s>>>(esrc[5], edst[5], ctxall, apctx, n);  // dff_src/dst
  }
  // tail GRU -> new_tail output slot
  {
    long long n = (long long)N_FLT * 128;
    gather_rows_k<<<nblk(n), 256, 0, s>>>(tail_states, tail_id, prevh, n, 128);
  }
  gemm(s, x_fl,  PRM(78), PRM(80), gi_t, N_FLT, 384, 256, 256, 256, 384, 1);
  gemm(s, prevh, PRM(79), PRM(81), gh_t, N_FLT, 384, 128, 128, 128, 384, 1);
  {
    long long n = (long long)N_FLT * 128;
    gru_gate_k<<<nblk(n), 256, 0, s>>>(gi_t, gh_t, prevh, o_tail, n, 128);
  }
  gemm(s, o_tail, PRM(82), PRM(83), tailctx, N_FLT, 256, 128, 128, 256, 256);

  // fusion: cat @ W done as 3 accumulating GEMMs over W's row blocks
  gemm(s, x_fl,    PRM(84),             PRM(85), ffacc, N_FLT, 256, 256, 256, 256, 256, 0, 0, 0);
  gemm(s, apctx,   PRM(84) + 256 * 256, nullptr, ffacc, N_FLT, 256, 256, 256, 256, 256, 0, 0, 1);
  gemm(s, tailctx, PRM(84) + 512 * 256, nullptr, ffacc, N_FLT, 256, 256, 256, 256, 256, 0, 0, 1);
  ln256_k<<<(N_FLT + 7) / 8, 256, 0, s>>>(ffacc, PRM(86), PRM(87), N_FLT, 1);   // LN + relu
  gemm(s, x_fl,    PRM(88),             PRM(89), fgacc, N_FLT, 256, 256, 256, 256, 256, 0, 0, 0);
  gemm(s, apctx,   PRM(88) + 256 * 256, nullptr, fgacc, N_FLT, 256, 256, 256, 256, 256, 0, 0, 1);
  gemm(s, tailctx, PRM(88) + 512 * 256, nullptr, fgacc, N_FLT, 256, 256, 256, 256, 256, 0, 0, 1);
  sigmoid_k<<<nblk((long long)N_FLT * 256), 256, 0, s>>>(fgacc, (long long)N_FLT * 256);
  fuse_out_k<<<nblk((long long)N_FLT * 256), 256, 0, s>>>(fgacc, ffacc, x_fl, flout, (long long)N_FLT * 256);

  // flight heads
  gemm(s, flout, PRM(96),  PRM(97),  f1, N_FLT, 256, 256, 256, 256, 256, 0, 1);
  gemm(s, f1,    PRM(98),  PRM(99),  f2, N_FLT, 128, 256, 256, 128, 128, 0, 1);
  gemm(s, f2,    PRM(100), PRM(101), zb, N_FLT, 1,   128, 128, 1,   1);
  gemm(s, flout, PRM(102), PRM(103), f1, N_FLT, 128, 256, 256, 128, 128, 0, 1);
  gemm(s, f1,    PRM(104), PRM(105), o_logits, N_FLT, 6, 128, 128, 6, 6);
  gemm(s, flout, PRM(106), PRM(107), f2, N_FLT, 128, 256, 256, 128, 128, 0, 1);
  gemm(s, f2,    PRM(108), PRM(109), mag, N_FLT, 2, 128, 128, 2, 2);
  final_k<<<nblk(N_FLT), 256, 0, s>>>(zb, o_logits, mag, o_flpred, o_flz, N_FLT);
  #undef PRM
}
